// GatedDeltaNet_70557722739260
// MI455X (gfx1250) — matrix operand order repaired and checked by reading, not compiled, not run on hardware
//
#include <hip/hip_runtime.h>
#include <hip/hip_bf16.h>

// ---------------- problem constants (match reference) ----------------
#define BB 2
#define TT 2048
#define DD 2048
#define HK 16
#define HV 32
#define DK 128
#define DV 128
#define KEY_DIM 2048
#define VALUE_DIM 4096
#define CONV_DIM 8192
#define KCONV 4
#define MROWS (BB * TT)        // 4096
constexpr float EPSF = 1e-6f;

// ---------------- types for WMMA ----------------
typedef __attribute__((ext_vector_type(16))) __bf16 v16bf;
typedef __attribute__((ext_vector_type(8)))  float  v8f;
typedef __attribute__((ext_vector_type(8)))  unsigned int v8u;

union FragAB { v16bf bf; v8u u; };

// ---------------- helpers ----------------
__device__ __forceinline__ unsigned short f2bf(float f) {
    unsigned int u = __float_as_uint(f);
    unsigned int r = u + 0x7FFFu + ((u >> 16) & 1u);   // round-to-nearest-even
    return (unsigned short)(r >> 16);
}

// ---------------- cast kernels ----------------
__global__ void cast_bf16_kernel(const float* __restrict__ in,
                                 unsigned short* __restrict__ out, long n) {
    long id = (long)blockIdx.x * blockDim.x + threadIdx.x;
    if (id < n) out[id] = f2bf(in[id]);
}

// in [K,N] fp32 -> out [N,K] bf16 (transposed, so GEMM B-fragments are contiguous)
__global__ void cast_bf16_T_kernel(const float* __restrict__ in,
                                   unsigned short* __restrict__ out, int K, int N) {
    long id = (long)blockIdx.x * blockDim.x + threadIdx.x;
    if (id >= (long)K * N) return;
    int k = (int)(id % K);
    int ncol = (int)(id / K);
    out[id] = f2bf(in[(long)k * N + ncol]);
}

// ---------------- bf16 WMMA GEMM: C[M,N] = A[M,K] * B[K,N] ----------------
// A row-major bf16, Bt = B transposed ([N,K] row-major bf16), C fp32.
// One wave -> 64x64 output tile (4x4 fragments, 16 WMMAs per K-step of 32);
// 8 waves per block. A-frags reused across 4 N-tiles, B-frags across 4 M-tiles.
__global__ __launch_bounds__(256) void gemm_bf16_wmma(
    const unsigned short* __restrict__ A,
    const unsigned short* __restrict__ Bt,
    float* __restrict__ C, int M, int N, int K)
{
    const int tid  = threadIdx.x;
    const int lane = tid & 31;
    const int wave = tid >> 5;
    const int tilesN = N >> 6;                 // 64-wide tiles
    const int wid = blockIdx.x * 8 + wave;
    const int tm = (wid / tilesN) << 6;
    const int tn = (wid % tilesN) << 6;
    const int r  = lane & 15;
    const bool lo = lane < 16;
    const int aOff = lo ? 0 : 16;   // contiguous k order (matches B): hi lanes hold K+16
    const int bOff = lo ? 0 : 16;   // contiguous k order (matches A): hi lanes hold K+16

    const unsigned short* Arow[4];
    const unsigned short* Bcol[4];
    #pragma unroll
    for (int i = 0; i < 4; ++i) {
        Arow[i] = A  + (size_t)(tm + 16 * i + r) * K;
        Bcol[i] = Bt + (size_t)(tn + 16 * i + r) * K;
    }

    v8f acc[4][4] = {};

    for (int kk = 0; kk < K; kk += 32) {
        FragAB a[4], b[4];
        #pragma unroll
        for (int i = 0; i < 4; ++i) {
            #pragma unroll
            for (int v = 0; v < 8; ++v) {
                const int ka = kk + aOff + 2 * v;
                const int kb = kk + bOff + 2 * v;
                a[i].u[v] = *(const unsigned int*)(Arow[i] + ka);
                b[i].u[v] = *(const unsigned int*)(Bcol[i] + kb);
            }
        }
        // pull next K-cacheline toward the WGP while the 16 WMMAs run
        if (kk + 64 < K) {
            #pragma unroll
            for (int i = 0; i < 4; ++i) {
                __builtin_prefetch(Arow[i] + kk + 64 + aOff, 0, 3);
                __builtin_prefetch(Bcol[i] + kk + 64 + bOff, 0, 3);
            }
        }
        #pragma unroll
        for (int i = 0; i < 4; ++i)
            #pragma unroll
            for (int j = 0; j < 4; ++j)
                acc[i][j] = __builtin_amdgcn_wmma_f32_16x16x32_bf16(
                    false, a[i].bf, false, b[j].bf,
                    (short)0, acc[i][j], false, false);
    }

    #pragma unroll
    for (int i = 0; i < 4; ++i)
        #pragma unroll
        for (int j = 0; j < 4; ++j) {
            const int col = tn + j * 16 + r;
            #pragma unroll
            for (int rv = 0; rv < 8; ++rv) {
                const int row = tm + i * 16 + rv + (lo ? 0 : 8);
                C[(size_t)row * N + col] = acc[i][j][rv];
            }
        }
}

// ---------------- small GEMV: out[M,32] = x[M,K] @ w[K,32] ----------------
__global__ void gemv32_kernel(const float* __restrict__ x,
                              const float* __restrict__ w,
                              float* __restrict__ out, int K) {
    const int id  = blockIdx.x * blockDim.x + threadIdx.x;
    const int row = id >> 5;
    const int col = id & 31;
    const float* xr = x + (size_t)row * K;
    float acc = 0.f;
    for (int k = 0; k < K; ++k) acc += xr[k] * w[k * 32 + col];
    out[id] = acc;
}

// ---------------- fused causal conv1d + SiLU + l2norm(q,k) + split ----------------
__global__ __launch_bounds__(256) void conv_norm_kernel(
    const float* __restrict__ mixed,   // [B,T,CONV_DIM] pre-activation
    const float* __restrict__ conv_w,  // [KCONV, CONV_DIM]
    float* __restrict__ qn,            // [B,T,HK,DK] normalized * DK^-0.5
    float* __restrict__ kn,            // [B,T,HK,DK] normalized
    float* __restrict__ vbuf,          // [B,T,VALUE_DIM]
    int Tlen)
{
    const int bt  = blockIdx.x;            // b*T + t
    const int b   = bt / Tlen;
    const int t   = bt % Tlen;
    const int tid = threadIdx.x;
    __shared__ float smem[CONV_DIM];
    __shared__ float redsm[256];
    __shared__ float norms[32];

    #pragma unroll
    for (int i = 0; i < CONV_DIM / 256; ++i) {
        const int c = tid + (i << 8);
        float acc = 0.f;
        #pragma unroll
        for (int k = 0; k < KCONV; ++k) {
            const int tt = t - (KCONV - 1) + k;
            if (tt >= 0)
                acc += mixed[((size_t)b * Tlen + tt) * CONV_DIM + c] *
                       conv_w[k * CONV_DIM + c];
        }
        smem[c] = acc / (1.f + __expf(-acc));   // SiLU
    }
    __syncthreads();

    // sum-of-squares for 32 heads (16 q + 16 k), 8 threads/head
    {
        const int head = tid >> 3;
        const int j    = tid & 7;
        const int base = (head < 16) ? head * DK : KEY_DIM + (head - 16) * DK;
        float s = 0.f;
        #pragma unroll
        for (int i2 = 0; i2 < 16; ++i2) {
            const float v = smem[base + j * 16 + i2];
            s += v * v;
        }
        redsm[tid] = s;
    }
    __syncthreads();
    if (tid < 32) {
        float s = 0.f;
        #pragma unroll
        for (int j = 0; j < 8; ++j) s += redsm[tid * 8 + j];
        norms[tid] = rsqrtf(s + EPSF);
    }
    __syncthreads();

    const float qscale = 0.08838834764831845f;  // DK^-0.5
    #pragma unroll
    for (int i = 0; i < KEY_DIM / 256; ++i) {
        const int c = tid + (i << 8);
        const int hh = c >> 7;
        const int d  = c & 127;
        qn[((size_t)bt * HK + hh) * DK + d] = smem[c] * norms[hh] * qscale;
        kn[((size_t)bt * HK + hh) * DK + d] = smem[KEY_DIM + c] * norms[16 + hh];
    }
    #pragma unroll
    for (int i = 0; i < VALUE_DIM / 256; ++i) {
        const int c = tid + (i << 8);
        vbuf[(size_t)bt * VALUE_DIM + c] = smem[2 * KEY_DIM + c];
    }
}

// ---------------- decay / beta gates ----------------
__global__ void gate_kernel(const float* __restrict__ a_raw,
                            const float* __restrict__ b_raw,
                            const float* __restrict__ dt_bias,
                            const float* __restrict__ A_log,
                            float* __restrict__ g, float* __restrict__ beta, int n) {
    const int id = blockIdx.x * blockDim.x + threadIdx.x;
    if (id >= n) return;
    const int h = id & (HV - 1);
    const float x  = a_raw[id] + dt_bias[h];
    const float sp = (x > 20.f) ? x : log1pf(__expf(x));
    g[id]    = -__expf(A_log[h]) * sp;
    beta[id] = 1.f / (1.f + __expf(-b_raw[id]));
}

// ---------------- gated delta-rule sequential scan ----------------
// One block per (batch, head). State 128x128 fp32 in VGPRs: 256 threads,
// thread (half,e) owns d in [half*64, half*64+64) of column e.
__global__ __launch_bounds__(256) void scan_kernel(
    const float* __restrict__ qn, const float* __restrict__ kn,
    const float* __restrict__ vbuf, const float* __restrict__ gbuf,
    const float* __restrict__ betabuf, float* __restrict__ core, int Tlen)
{
    const int b    = blockIdx.x / HV;
    const int h    = blockIdx.x % HV;
    const int hk   = h >> 1;                 // GQA: repeat factor 2
    const int tid  = threadIdx.x;
    const int e    = tid & 127;
    const int half = tid >> 7;
    const int dbase = half * 64;

    __shared__ float sK[128], sQ[128], sV[128], sDelta[128], red[256];

    float st[64];
    #pragma unroll
    for (int i = 0; i < 64; ++i) st[i] = 0.f;

    for (int t = 0; t < Tlen; ++t) {
        const size_t bt = (size_t)b * Tlen + t;
        if (tid < 128) {
            sK[tid] = kn[(bt * HK + hk) * DK + tid];
            sV[tid] = vbuf[bt * VALUE_DIM + h * DV + tid];
        } else {
            sQ[tid - 128] = qn[(bt * HK + hk) * DK + (tid - 128)];
        }
        const float g     = gbuf[bt * HV + h];
        const float betav = betabuf[bt * HV + h];
        const float decay = __expf(g);
        __syncthreads();

        // kv_mem = k . (decay * S_old)  (decay factored out of the dot)
        float p = 0.f;
        #pragma unroll
        for (int i = 0; i < 64; ++i) p += sK[dbase + i] * st[i];
        red[tid] = p;
        __syncthreads();
        if (half == 0) {
            const float kv = decay * (red[e] + red[128 + e]);
            sDelta[e] = (sV[e] - kv) * betav;
        }
        __syncthreads();

        // state update + out = q . S_new
        const float delta = sDelta[e];
        float po = 0.f;
        #pragma unroll
        for (int i = 0; i < 64; ++i) {
            const float s2 = st[i] * decay + sK[dbase + i] * delta;
            st[i] = s2;
            po += sQ[dbase + i] * s2;
        }
        red[tid] = po;
        __syncthreads();
        if (half == 0)
            core[(bt * HV + h) * DV + e] = red[e] + red[128 + e];
        // red reuse next iter is guarded by the post-load __syncthreads()
    }
}

// ---------------- gated RMS norm -> bf16 for output GEMM ----------------
__global__ __launch_bounds__(128) void rmsnorm_gate_kernel(
    const float* __restrict__ core, const float* __restrict__ z,
    const float* __restrict__ norm_w, unsigned short* __restrict__ attn)
{
    const int bh = blockIdx.x;           // (b*T+t)*HV + h
    const int e  = threadIdx.x;
    __shared__ float rs[128];
    const float c = core[(size_t)bh * DV + e];
    rs[e] = c * c;
    __syncthreads();
    for (int s = 64; s > 0; s >>= 1) {
        if (e < s) rs[e] += rs[e + s];
        __syncthreads();
    }
    const float rinv = rsqrtf(rs[0] * (1.f / DV) + EPSF);
    const size_t bt = bh / HV;
    const int h = bh % HV;
    const float zv = z[bt * VALUE_DIM + h * DV + e];
    const float y  = norm_w[e] * c * rinv * (zv / (1.f + __expf(-zv)));
    attn[(size_t)bh * DV + e] = f2bf(y);
}

// ---------------- host-side launcher ----------------
extern "C" void kernel_launch(void* const* d_in, const int* in_sizes, int n_in,
                              void* d_out, int out_size, void* d_ws, size_t ws_size,
                              hipStream_t stream) {
    (void)in_sizes; (void)n_in; (void)out_size; (void)ws_size;
    const float* x       = (const float*)d_in[0];
    const float* w_qkv   = (const float*)d_in[1];
    const float* w_z     = (const float*)d_in[2];
    const float* w_b     = (const float*)d_in[3];
    const float* w_a     = (const float*)d_in[4];
    const float* conv_w  = (const float*)d_in[5];
    const float* dt_bias = (const float*)d_in[6];
    const float* A_log   = (const float*)d_in[7];
    const float* norm_w  = (const float*)d_in[8];
    const float* w_out   = (const float*)d_in[9];
    float* out = (float*)d_out;
    char*  ws  = (char*)d_ws;

    // workspace layout (256B-aligned); core aliases mixed, attn aliases wqkv_bf16
    size_t off = 0;
    auto alloc = [&](size_t bytes) { size_t o = off; off += (bytes + 255) & ~(size_t)255; return o; };
    const size_t o_xb    = alloc((size_t)MROWS * DD * 2);          // x bf16
    const size_t o_wqkvb = alloc((size_t)DD * CONV_DIM * 2);       // w_qkv^T bf16
    const size_t o_wzb   = alloc((size_t)DD * VALUE_DIM * 2);      // w_z^T bf16
    const size_t o_woutb = alloc((size_t)VALUE_DIM * DD * 2);      // w_out^T bf16
    const size_t o_mixed = alloc((size_t)MROWS * CONV_DIM * 4);    // pre-conv projections
    const size_t o_z     = alloc((size_t)MROWS * VALUE_DIM * 4);
    const size_t o_braw  = alloc((size_t)MROWS * HV * 4);
    const size_t o_araw  = alloc((size_t)MROWS * HV * 4);
    const size_t o_qn    = alloc((size_t)MROWS * HK * DK * 4);
    const size_t o_kn    = alloc((size_t)MROWS * HK * DK * 4);
    const size_t o_v     = alloc((size_t)MROWS * VALUE_DIM * 4);
    const size_t o_g     = alloc((size_t)MROWS * HV * 4);
    const size_t o_beta  = alloc((size_t)MROWS * HV * 4);
    const size_t o_core  = o_mixed;     // mixed dead after conv_norm_kernel
    const size_t o_attn  = o_wqkvb;     // wqkv bf16 dead after first GEMM

    unsigned short* xb    = (unsigned short*)(ws + o_xb);
    unsigned short* wqkvb = (unsigned short*)(ws + o_wqkvb);
    unsigned short* wzb   = (unsigned short*)(ws + o_wzb);
    unsigned short* woutb = (unsigned short*)(ws + o_woutb);
    float* mixed = (float*)(ws + o_mixed);
    float* zbuf  = (float*)(ws + o_z);
    float* braw  = (float*)(ws + o_braw);
    float* araw  = (float*)(ws + o_araw);
    float* qn    = (float*)(ws + o_qn);
    float* kn    = (float*)(ws + o_kn);
    float* vbuf  = (float*)(ws + o_v);
    float* gbuf  = (float*)(ws + o_g);
    float* bbuf  = (float*)(ws + o_beta);
    float* core  = (float*)(ws + o_core);
    unsigned short* attn = (unsigned short*)(ws + o_attn);

    // 1) casts to bf16 (weights transposed for contiguous WMMA B-fragments)
    {
        long n = (long)MROWS * DD;
        cast_bf16_kernel<<<(unsigned)((n + 255) / 256), 256, 0, stream>>>(x, xb, n);
    }
    {
        long n = (long)DD * CONV_DIM;
        cast_bf16_T_kernel<<<(unsigned)((n + 255) / 256), 256, 0, stream>>>(w_qkv, wqkvb, DD, CONV_DIM);
    }
    {
        long n = (long)DD * VALUE_DIM;
        cast_bf16_T_kernel<<<(unsigned)((n + 255) / 256), 256, 0, stream>>>(w_z, wzb, DD, VALUE_DIM);
    }
    {
        long n = (long)VALUE_DIM * DD;
        cast_bf16_T_kernel<<<(unsigned)((n + 255) / 256), 256, 0, stream>>>(w_out, woutb, VALUE_DIM, DD);
    }

    // 2) projection GEMMs (WMMA bf16 -> f32), 64x64 tile per wave
    gemm_bf16_wmma<<<(MROWS / 64) * (CONV_DIM / 64) / 8, 256, 0, stream>>>(
        xb, wqkvb, mixed, MROWS, CONV_DIM, DD);
    gemm_bf16_wmma<<<(MROWS / 64) * (VALUE_DIM / 64) / 8, 256, 0, stream>>>(
        xb, wzb, zbuf, MROWS, VALUE_DIM, DD);

    // 3) small gate projections (fp32)
    gemv32_kernel<<<(MROWS * HV) / 256, 256, 0, stream>>>(x, w_b, braw, DD);
    gemv32_kernel<<<(MROWS * HV) / 256, 256, 0, stream>>>(x, w_a, araw, DD);

    // 4) conv + SiLU + l2norm + split
    conv_norm_kernel<<<MROWS, 256, 0, stream>>>(mixed, conv_w, qn, kn, vbuf, TT);

    // 5) decay/beta gates
    gate_kernel<<<(MROWS * HV) / 256, 256, 0, stream>>>(araw, braw, dt_bias, A_log, gbuf, bbuf, MROWS * HV);

    // 6) sequential gated delta-rule scan (state in VGPRs)
    scan_kernel<<<BB * HV, 256, 0, stream>>>(qn, kn, vbuf, gbuf, bbuf, core, TT);

    // 7) gated RMS norm -> bf16
    rmsnorm_gate_kernel<<<MROWS * HV, 128, 0, stream>>>(core, zbuf, norm_w, attn);

    // 8) output GEMM (WMMA)
    gemm_bf16_wmma<<<(MROWS / 64) * (DD / 64) / 8, 256, 0, stream>>>(
        attn, woutb, out, MROWS, DD, VALUE_DIM);
}
